// TransformerSubsetAggregator_87969520156961
// MI455X (gfx1250) — matrix operand order repaired and checked by reading, not compiled, not run on hardware
//
#include <hip/hip_runtime.h>
#include <math.h>

// Sizes from the reference
#define D_   256
#define FF_  512
#define NH_  4
#define HD_  64
#define KTOK 8
#define LSEQ 64
#define NC_  32768
#define NB_  4096
#define FPAD 288   // 257 padded up to a multiple of 32 for WMMA K loops

typedef __attribute__((ext_vector_type(16))) __bf16 v16bf;
typedef __attribute__((ext_vector_type(8)))  __bf16 v8bf;
typedef __attribute__((ext_vector_type(8)))  float  v8f;

__device__ __forceinline__ __bf16 f2bf(float f) {
  unsigned u = __float_as_uint(f);
  u = (u + 0x7FFFu + ((u >> 16) & 1u)) >> 16;   // round-to-nearest-even
  unsigned short s = (unsigned short)u;
  __bf16 b; __builtin_memcpy(&b, &s, 2); return b;
}
__device__ __forceinline__ float bf2f(__bf16 b) {
  unsigned short s; __builtin_memcpy(&s, &b, 2);
  unsigned u = ((unsigned)s) << 16; return __uint_as_float(u);
}
__device__ __forceinline__ float gelu_exact(float x) {
  return 0.5f * x * (1.0f + erff(x * 0.70710678118654752f));
}

// ---- 16xN WMMA row-tile: acc += A(16 x klen, bf16 in LDS/global) * B^T ----
// A: row-major, stride a_stride. B: weight rows [n][k] row-major, pre-offset
// to the 16-column base; stride b_stride. Lane layouts per CDNA5 ISA.
__device__ __forceinline__ v8f wmma_rowtile(const __bf16* A, int a_stride,
                                            const __bf16* B, int b_stride,
                                            int klen, int lane, v8f acc) {
  const int ar = lane & 15;
  const int hi = lane >> 4;
  for (int k0 = 0; k0 < klen; k0 += 32) {
    v8bf a1 = *(const v8bf*)(A + ar * a_stride + k0 + hi * 8);
    v8bf a2 = *(const v8bf*)(A + ar * a_stride + k0 + hi * 8 + 16);
    v16bf av = __builtin_shufflevector(a1, a2, 0,1,2,3,4,5,6,7,8,9,10,11,12,13,14,15);
    v8bf b1 = *(const v8bf*)(B + ar * b_stride + k0 + hi * 8);
    v8bf b2 = *(const v8bf*)(B + ar * b_stride + k0 + hi * 8 + 16);
    v16bf bv = __builtin_shufflevector(b1, b2, 0,1,2,3,4,5,6,7,8,9,10,11,12,13,14,15);
    acc = __builtin_amdgcn_wmma_f32_16x16x32_bf16(false, av, false, bv,
                                                  (short)0, acc, false, false);
  }
  return acc;
}

__device__ __forceinline__ void layernorm16(const float* xf, __bf16* dst, int dst_stride,
                                            const float* w, const float* b, int lane) {
  const int r = lane & 15, hi = lane >> 4;
  const float* row = xf + r * D_;
  const int c0 = hi * 128;
  float s = 0.f;
  for (int c = c0; c < c0 + 128; ++c) s += row[c];
  s += __shfl_xor(s, 16, 32);
  const float mu = s * (1.0f / 256.0f);
  float v2 = 0.f;
  for (int c = c0; c < c0 + 128; ++c) { float d = row[c] - mu; v2 += d * d; }
  v2 += __shfl_xor(v2, 16, 32);
  const float rs = rsqrtf(v2 * (1.0f / 256.0f) + 1e-5f);
  for (int c = c0; c < c0 + 128; ++c)
    dst[r * dst_stride + c] = f2bf((row[c] - mu) * rs * w[c] + b[c]);
}

// ---------------- weight prep ----------------
__global__ void f32_to_bf16_kernel(const float* __restrict__ src,
                                   __bf16* __restrict__ dst, int n) {
  for (int i = blockIdx.x * blockDim.x + threadIdx.x; i < n; i += gridDim.x * blockDim.x)
    dst[i] = f2bf(src[i]);
}

__global__ void pad_o1_kernel(const float* __restrict__ src, __bf16* __restrict__ dst) {
  // (512 x 257) -> (512 x 288), zero padded K
  for (int i = blockIdx.x * blockDim.x + threadIdx.x; i < FF_ * FPAD; i += gridDim.x * blockDim.x) {
    int n = i / FPAD, k = i - n * FPAD;
    dst[i] = (k < D_ + 1) ? f2bf(src[n * (D_ + 1) + k]) : f2bf(0.f);
  }
}

// ---------------- selection: first <=8 set bits of mask, ascending ----------------
__global__ void select_kernel(const unsigned char* __restrict__ mask,
                              int* __restrict__ selidx, int* __restrict__ attmask,
                              int* __restrict__ npick) {
  int c = blockIdx.x * blockDim.x + threadIdx.x;
  if (c >= NC_) return;
  const unsigned char* m = mask + (size_t)c * LSEQ;
  int sel[KTOK];
#pragma unroll
  for (int k = 0; k < KTOK; ++k) sel[k] = -1;
  int cnt = 0;
  for (int pos = 0; pos < LSEQ && cnt < KTOK; ++pos)
    if (m[pos]) sel[cnt++] = pos;
#pragma unroll
  for (int k = 0; k < KTOK; ++k) selidx[c * KTOK + k] = sel[k];
  npick[c] = cnt;
  attmask[c] = cnt ? ((1 << cnt) - 1) : 1;   // kpm[0]=False fix for empty clusters
}

// ---------------- fused transformer block: 2 clusters (16 rows) per wave ----------------
__global__ void __launch_bounds__(32)
fused_block_kernel(const float* __restrict__ vsrc,
                   const int* __restrict__ batch_idx,
                   const float* __restrict__ count,
                   const float* __restrict__ pos_embed,
                   const float* __restrict__ ln1_w, const float* __restrict__ ln1_b,
                   const float* __restrict__ in_proj_b,
                   const float* __restrict__ out_proj_b,
                   const float* __restrict__ ln2_w, const float* __restrict__ ln2_b,
                   const float* __restrict__ ff1_b, const float* __restrict__ ff2_b,
                   const __bf16* __restrict__ wq,   // 768 x 256
                   const __bf16* __restrict__ wo,   // 256 x 256
                   const __bf16* __restrict__ w1,   // 512 x 256
                   const __bf16* __restrict__ w2,   // 256 x 512
                   const int* __restrict__ selidx,
                   const int* __restrict__ attmask,
                   const int* __restrict__ npick,
                   __bf16* __restrict__ feat) {
  __shared__ float  xf[16 * 256];     // residual stream, f32 (16 KB)
  __shared__ __bf16 ab[16 * 512];     // A operands: LN1/ao/LN2 (16 KB)
  __shared__ __bf16 qkvb[16 * 768];   // qkv, later gelu(ff1) (24 KB)

  const int lane  = threadIdx.x;
  const int cbase = blockIdx.x * 2;

  // ---- gather: x = packed*picked + pos_embed ----
  for (int i = lane; i < 16 * 256; i += 32) {
    const int r = i >> 8, col = i & 255;
    const int c = r >> 3, slot = r & 7;
    const int cg = cbase + c;
    const int idx = selidx[cg * KTOK + slot];
    float val = pos_embed[slot * D_ + col];
    if (idx >= 0) {
      const int b = batch_idx[cg];
      val += vsrc[((size_t)b * LSEQ + idx) * D_ + col];
    }
    xf[i] = val;
  }
  __syncthreads();

  // ---- LN1 ----
  layernorm16(xf, ab, 512, ln1_w, ln1_b, lane);
  __syncthreads();

  // ---- QKV: 16x256 @ 256x768 ----
  for (int nt = 0; nt < 48; ++nt) {
    const int nb = nt * 16;
    v8f acc = {0.f,0.f,0.f,0.f,0.f,0.f,0.f,0.f};
    acc = wmma_rowtile(ab, 512, wq + (size_t)nb * D_, D_, D_, lane, acc);
    const int n = nb + (lane & 15);
    const float bias = in_proj_b[n];
#pragma unroll
    for (int j = 0; j < 8; ++j) {
      const int m = j + (lane >> 4) * 8;
      qkvb[m * 768 + n] = f2bf(acc[j] + bias);
    }
  }
  __syncthreads();

  // ---- attention: per (cluster, head, q-row), masked softmax over 8 slots ----
  const int am0 = attmask[cbase], am1 = attmask[cbase + 1];
#pragma unroll
  for (int t = 0; t < 2; ++t) {
    const int combo = lane + 32 * t;
    const int c  = combo >> 5;
    const int h  = (combo >> 3) & 3;
    const int qi = combo & 7;
    const int am = c ? am1 : am0;
    const int d0 = h * HD_;
    const int qrow = c * 8 + qi;
    float sc[KTOK];
#pragma unroll
    for (int ki = 0; ki < KTOK; ++ki) {
      float s = 0.f;
      const __bf16* qp = &qkvb[qrow * 768 + d0];
      const __bf16* kp = &qkvb[(c * 8 + ki) * 768 + D_ + d0];
      for (int d = 0; d < HD_; ++d) s += bf2f(qp[d]) * bf2f(kp[d]);
      sc[ki] = s * 0.125f;   // 1/sqrt(64)
    }
    float mx = -3.0e38f;
#pragma unroll
    for (int ki = 0; ki < KTOK; ++ki) if ((am >> ki) & 1) mx = fmaxf(mx, sc[ki]);
    float wsum = 0.f, w8[KTOK];
#pragma unroll
    for (int ki = 0; ki < KTOK; ++ki) {
      w8[ki] = ((am >> ki) & 1) ? __expf(sc[ki] - mx) : 0.f;
      wsum += w8[ki];
    }
    const float inv = 1.f / wsum;
    for (int d = 0; d < HD_; ++d) {
      float o = 0.f;
#pragma unroll
      for (int ki = 0; ki < KTOK; ++ki)
        o += w8[ki] * bf2f(qkvb[(c * 8 + ki) * 768 + 2 * D_ + d0 + d]);
      ab[qrow * 512 + d0 + d] = f2bf(o * inv);   // ao, heads re-concatenated
    }
  }
  __syncthreads();

  // ---- out-proj + residual ----
  for (int nt = 0; nt < 16; ++nt) {
    const int nb = nt * 16;
    v8f acc = {0.f,0.f,0.f,0.f,0.f,0.f,0.f,0.f};
    acc = wmma_rowtile(ab, 512, wo + (size_t)nb * D_, D_, D_, lane, acc);
    const int n = nb + (lane & 15);
    const float bias = out_proj_b[n];
#pragma unroll
    for (int j = 0; j < 8; ++j) {
      const int m = j + (lane >> 4) * 8;
      xf[m * 256 + n] += acc[j] + bias;
    }
  }
  __syncthreads();

  // ---- LN2 ----
  layernorm16(xf, ab, 512, ln2_w, ln2_b, lane);
  __syncthreads();

  // ---- FF1 + exact GELU ----
  for (int nt = 0; nt < 32; ++nt) {
    const int nb = nt * 16;
    v8f acc = {0.f,0.f,0.f,0.f,0.f,0.f,0.f,0.f};
    acc = wmma_rowtile(ab, 512, w1 + (size_t)nb * D_, D_, D_, lane, acc);
    const int n = nb + (lane & 15);
    const float bias = ff1_b[n];
#pragma unroll
    for (int j = 0; j < 8; ++j) {
      const int m = j + (lane >> 4) * 8;
      qkvb[m * 768 + n] = f2bf(gelu_exact(acc[j] + bias));
    }
  }
  __syncthreads();

  // ---- FF2 + residual ----
  for (int nt = 0; nt < 16; ++nt) {
    const int nb = nt * 16;
    v8f acc = {0.f,0.f,0.f,0.f,0.f,0.f,0.f,0.f};
    acc = wmma_rowtile(qkvb, 768, w2 + (size_t)nb * FF_, FF_, FF_, lane, acc);
    const int n = nb + (lane & 15);
    const float bias = ff2_b[n];
#pragma unroll
    for (int j = 0; j < 8; ++j) {
      const int m = j + (lane >> 4) * 8;
      xf[m * 256 + n] += acc[j] + bias;
    }
  }
  __syncthreads();

  // ---- masked mean pool -> feat (bf16, padded to 288) ----
#pragma unroll
  for (int c = 0; c < 2; ++c) {
    const int cg = cbase + c;
    const int np = npick[cg];
    const int am = c ? am1 : am0;
    const float invn = (np > 0) ? (1.0f / (float)np) : 0.f;
    for (int d = lane; d < D_; d += 32) {
      float s = 0.f;
      if (np > 0) {
#pragma unroll
        for (int k = 0; k < KTOK; ++k)
          if ((am >> k) & 1) s += xf[(c * 8 + k) * 256 + d];
        s *= invn;
      }
      feat[(size_t)cg * FPAD + d] = f2bf(s);
    }
    if (lane == 0) feat[(size_t)cg * FPAD + D_] = f2bf(log1pf(count[cg]));
    for (int d = D_ + 1 + lane; d < FPAD; d += 32)
      feat[(size_t)cg * FPAD + d] = f2bf(0.f);
  }
}

// ---------------- head MLP: 16 clusters per wave, pure WMMA ----------------
__global__ void __launch_bounds__(32)
head_kernel(const __bf16* __restrict__ feat,   // NC x 288
            const __bf16* __restrict__ wo1,    // 512 x 288
            const float*  __restrict__ o1_b,
            const __bf16* __restrict__ wo2,    // 256 x 512
            const float*  __restrict__ o2_b,
            float* __restrict__ out) {
  __shared__ __bf16 gb[16 * 512];
  const int lane  = threadIdx.x;
  const int mbase = blockIdx.x * 16;

  // GEMM1: 16x288 @ 288x512, A straight from global (L2-resident feat)
  const __bf16* A = feat + (size_t)mbase * FPAD;
  for (int nt = 0; nt < 32; ++nt) {
    const int nb = nt * 16;
    v8f acc = {0.f,0.f,0.f,0.f,0.f,0.f,0.f,0.f};
    acc = wmma_rowtile(A, FPAD, wo1 + (size_t)nb * FPAD, FPAD, FPAD, lane, acc);
    const int n = nb + (lane & 15);
    const float bias = o1_b[n];
#pragma unroll
    for (int j = 0; j < 8; ++j) {
      const int m = j + (lane >> 4) * 8;
      gb[m * 512 + n] = f2bf(gelu_exact(acc[j] + bias));
    }
  }
  __syncthreads();

  // GEMM2: 16x512 @ 512x256 -> f32 output
  for (int nt = 0; nt < 16; ++nt) {
    const int nb = nt * 16;
    v8f acc = {0.f,0.f,0.f,0.f,0.f,0.f,0.f,0.f};
    acc = wmma_rowtile(gb, 512, wo2 + (size_t)nb * FF_, FF_, FF_, lane, acc);
    const int n = nb + (lane & 15);
    const float bias = o2_b[n];
#pragma unroll
    for (int j = 0; j < 8; ++j) {
      const int m = j + (lane >> 4) * 8;
      out[(size_t)(mbase + m) * D_ + n] = acc[j] + bias;
    }
  }
}

extern "C" void kernel_launch(void* const* d_in, const int* in_sizes, int n_in,
                              void* d_out, int out_size, void* d_ws, size_t ws_size,
                              hipStream_t stream) {
  (void)in_sizes; (void)n_in; (void)out_size; (void)ws_size;

  const float* vsrc       = (const float*)d_in[0];
  const int*   batch_idx  = (const int*)d_in[1];
  const unsigned char* mask = (const unsigned char*)d_in[2];   // bool = 1 byte
  const float* count      = (const float*)d_in[3];
  const float* pos_embed  = (const float*)d_in[4];
  const float* ln1_w      = (const float*)d_in[5];
  const float* ln1_b      = (const float*)d_in[6];
  const float* in_proj_w  = (const float*)d_in[7];
  const float* in_proj_b  = (const float*)d_in[8];
  const float* out_proj_w = (const float*)d_in[9];
  const float* out_proj_b = (const float*)d_in[10];
  const float* ln2_w      = (const float*)d_in[11];
  const float* ln2_b      = (const float*)d_in[12];
  const float* ff1_w      = (const float*)d_in[13];
  const float* ff1_b      = (const float*)d_in[14];
  const float* ff2_w      = (const float*)d_in[15];
  const float* ff2_b      = (const float*)d_in[16];
  const float* o1_w       = (const float*)d_in[17];
  const float* o1_b       = (const float*)d_in[18];
  const float* o2_w       = (const float*)d_in[19];
  const float* o2_b       = (const float*)d_in[20];
  float* out = (float*)d_out;

  // workspace carve (256B aligned)
  uintptr_t base = (uintptr_t)d_ws;
  size_t off = 0;
  auto carve = [&](size_t bytes) -> void* {
    void* p = (void*)(base + off);
    off += (bytes + 255) & ~(size_t)255;
    return p;
  };
  int*    selidx = (int*)carve((size_t)NC_ * KTOK * 4);
  int*    attm   = (int*)carve((size_t)NC_ * 4);
  int*    np     = (int*)carve((size_t)NC_ * 4);
  __bf16* wq     = (__bf16*)carve((size_t)3 * D_ * D_ * 2);
  __bf16* wo     = (__bf16*)carve((size_t)D_ * D_ * 2);
  __bf16* w1     = (__bf16*)carve((size_t)FF_ * D_ * 2);
  __bf16* w2     = (__bf16*)carve((size_t)D_ * FF_ * 2);
  __bf16* wo1    = (__bf16*)carve((size_t)FF_ * FPAD * 2);
  __bf16* wo2    = (__bf16*)carve((size_t)D_ * FF_ * 2);
  __bf16* feat   = (__bf16*)carve((size_t)NC_ * FPAD * 2);

  // weight conversion (f32 -> bf16, layouts already [N][K] row-major)
  f32_to_bf16_kernel<<<(3 * D_ * D_ + 255) / 256, 256, 0, stream>>>(in_proj_w, wq, 3 * D_ * D_);
  f32_to_bf16_kernel<<<(D_ * D_ + 255) / 256, 256, 0, stream>>>(out_proj_w, wo, D_ * D_);
  f32_to_bf16_kernel<<<(FF_ * D_ + 255) / 256, 256, 0, stream>>>(ff1_w, w1, FF_ * D_);
  f32_to_bf16_kernel<<<(D_ * FF_ + 255) / 256, 256, 0, stream>>>(ff2_w, w2, D_ * FF_);
  f32_to_bf16_kernel<<<(D_ * FF_ + 255) / 256, 256, 0, stream>>>(o2_w, wo2, D_ * FF_);
  pad_o1_kernel<<<(FF_ * FPAD + 255) / 256, 256, 0, stream>>>(o1_w, wo1);

  // token selection
  select_kernel<<<NC_ / 256, 256, 0, stream>>>(mask, selidx, attm, np);

  // fused transformer block: 2 clusters per wave32 workgroup
  fused_block_kernel<<<NC_ / 2, 32, 0, stream>>>(
      vsrc, batch_idx, count, pos_embed,
      ln1_w, ln1_b, in_proj_b, out_proj_b, ln2_w, ln2_b, ff1_b, ff2_b,
      wq, wo, w1, w2, selidx, attm, np, feat);

  // head MLP: 16 clusters per wave32 workgroup
  head_kernel<<<NC_ / 16, 32, 0, stream>>>(feat, wo1, o1_b, wo2, o2_b, out);
}
